// GRU_86672440033876
// MI455X (gfx1250) — compile-verified, matrix operand order checked by reading, not run on hardware
//
#include <hip/hip_runtime.h>

#define B_   64
#define T_   1024
#define H_   128
#define G3   384          // 3*H
#define KIN  96           // INPUT=86 padded to 96
#define NENV 512
#define MTOT 65536        // B*T

typedef __attribute__((ext_vector_type(16))) _Float16 v16h;
typedef __attribute__((ext_vector_type(8)))  _Float16 v8h;
typedef __attribute__((ext_vector_type(8)))  float    v8f;

// Load a 16x32 f16 A-fragment (or B^T fragment) from a row-major matrix.
// Per CDNA5 ISA layout: lane%16 = row, lane/16 selects K-half; per lane the
// 16 halfs are two contiguous groups of 8: K in [kh*8, kh*8+8) and [16+kh*8, ...).
// Caller pre-offsets `base` by the K origin (multiple of 32 halfs).
__device__ __forceinline__ v16h load_frag(const _Float16* base, int ld, int row0) {
  int lane = threadIdx.x & 31;
  int r = lane & 15, kh = lane >> 4;
  const _Float16* p = base + (size_t)(row0 + r) * ld + kh * 8;
  v8h lo = *(const v8h*)(p);
  v8h hi = *(const v8h*)(p + 16);
  v16h o;
#pragma unroll
  for (int i = 0; i < 8; ++i) { o[i] = lo[i]; o[i + 8] = hi[i]; }
  return o;
}

__device__ __forceinline__ v8f wmma_f16(v16h a, v16h b, v8f c) {
  return __builtin_amdgcn_wmma_f32_16x16x32_f16(false, a, false, b, (short)0, c,
                                                false, false);
}

__device__ __forceinline__ float sigmoidf_(float x) {
  return 1.f / (1.f + __expf(-x));
}

// Async DMA copy of 32 bytes (per lane) from global to LDS, tracked by ASYNCcnt.
__device__ __forceinline__ void async_copy32(_Float16* lds_dst,
                                             const _Float16* gsrc) {
  unsigned int dst = (unsigned int)(unsigned long long)lds_dst;  // low 32 = LDS off
  asm volatile("global_load_async_to_lds_b128 %0, %1, off"
               :: "v"(dst), "v"(gsrc) : "memory");
  asm volatile("global_load_async_to_lds_b128 %0, %1, off offset:16"
               :: "v"(dst), "v"(gsrc) : "memory");
}

__device__ __forceinline__ void wait_async0() {
  asm volatile("s_wait_asynccnt 0" ::: "memory");
}

// ---------------------------------------------------------------- pack / cast
__global__ void k_pack(const float* __restrict__ x, const float* __restrict__ W_ih,
                       const float* __restrict__ W_hh, const float* __restrict__ W1,
                       _Float16* __restrict__ xh, _Float16* __restrict__ Wih_h,
                       _Float16* __restrict__ Whh_h, _Float16* __restrict__ W1h) {
  size_t stride = (size_t)gridDim.x * blockDim.x;
  size_t tid = (size_t)blockIdx.x * blockDim.x + threadIdx.x;
  const size_t nxh = (size_t)T_ * B_ * KIN;   // (T,B,96) from x (B,T,86)
  for (size_t i = tid; i < nxh; i += stride) {
    int k = (int)(i % KIN);
    size_t tb = i / KIN;
    int b = (int)(tb % B_);
    int t = (int)(tb / B_);
    float v = (k < 86) ? x[((size_t)b * T_ + t) * 86 + k] : 0.f;
    xh[i] = (_Float16)v;
  }
  for (size_t i = tid; i < (size_t)G3 * KIN; i += stride) {
    int k = (int)(i % KIN);
    int n = (int)(i / KIN);
    Wih_h[i] = (_Float16)((k < 86) ? W_ih[(size_t)n * 86 + k] : 0.f);
  }
  for (size_t i = tid; i < (size_t)G3 * H_; i += stride)
    Whh_h[i] = (_Float16)W_hh[i];
  for (size_t i = tid; i < (size_t)NENV * 2 * H_; i += stride)
    W1h[i] = (_Float16)W1[i];
}

// ------------------------------------------------- Gi = x @ W_ih^T + b_ih
// One wave per 16x16 output tile; rows ordered (t*B + b).
__global__ void __launch_bounds__(256) k_gemm_gi(
    const _Float16* __restrict__ xh, const _Float16* __restrict__ Wih_h,
    const float* __restrict__ b_ih, float* __restrict__ Gi) {
  int wave = (blockIdx.x << 3) + (threadIdx.x >> 5);
  int tm = wave / (G3 / 16), tn = wave % (G3 / 16);
  int lane = threadIdx.x & 31;
  v8f acc = {};
#pragma unroll
  for (int s = 0; s < 3; ++s) {          // K = 96 = 3 * 32
    v16h a = load_frag(xh + s * 32, KIN, tm * 16);
    v16h b = load_frag(Wih_h + s * 32, KIN, tn * 16);
    acc = wmma_f16(a, b, acc);
  }
  int col = tn * 16 + (lane & 15);
  float bias = b_ih[col];
  int rbase = (lane >> 4) * 8;
#pragma unroll
  for (int r = 0; r < 8; ++r) {
    int row = tm * 16 + rbase + r;
    Gi[(size_t)row * G3 + col] = acc[r] + bias;
  }
}

// -------------------------------------------------------------- GRU recurrence
// 4 independent blocks of 16 batch rows. W_hh B-fragments pinned in VGPRs.
__global__ void __launch_bounds__(256) k_gru(
    const _Float16* __restrict__ Whh_h, const float* __restrict__ Gi,
    const float* __restrict__ h0, const float* __restrict__ b_hh,
    float* __restrict__ states_f, _Float16* __restrict__ states_h) {
  __shared__ __align__(16) _Float16 h16[16 * H_];
  __shared__ float h32[16 * H_];
  __shared__ float gh[16 * G3];
  int tid = threadIdx.x;
  int wave = tid >> 5, lane = tid & 31;
  int B0 = blockIdx.x * 16;

  v16h bf[3][4];                        // 3 N-tiles x 4 K-steps, loop-invariant
#pragma unroll
  for (int j = 0; j < 3; ++j)
#pragma unroll
    for (int s = 0; s < 4; ++s)
      bf[j][s] = load_frag(Whh_h + s * 32, H_, (wave * 3 + j) * 16);

  for (int e = tid; e < 16 * H_; e += 256) {
    float v = h0[(size_t)(B0 + e / H_) * H_ + (e % H_)];
    h32[e] = v;
    h16[e] = (_Float16)v;
  }
  __syncthreads();

  for (int t = 0; t < T_; ++t) {
    v16h a[4];
#pragma unroll
    for (int s = 0; s < 4; ++s) a[s] = load_frag(h16 + s * 32, H_, 0);
#pragma unroll
    for (int j = 0; j < 3; ++j) {
      v8f acc = {};
#pragma unroll
      for (int s = 0; s < 4; ++s) acc = wmma_f16(a[s], bf[j][s], acc);
      int col = (wave * 3 + j) * 16 + (lane & 15);
      int rbase = (lane >> 4) * 8;
#pragma unroll
      for (int r = 0; r < 8; ++r) gh[(rbase + r) * G3 + col] = acc[r];
    }
    __syncthreads();
    for (int e = tid; e < 16 * H_; e += 256) {
      int lb = e / H_, h = e % H_;
      const float* gi = Gi + ((size_t)t * B_ + (B0 + lb)) * G3;
      float ghr = gh[lb * G3 + h] + b_hh[h];
      float ghz = gh[lb * G3 + H_ + h] + b_hh[H_ + h];
      float ghn = gh[lb * G3 + 2 * H_ + h] + b_hh[2 * H_ + h];
      float r = sigmoidf_(gi[h] + ghr);
      float z = sigmoidf_(gi[H_ + h] + ghz);
      float n = tanhf(gi[2 * H_ + h] + r * ghn);
      float hn = (1.f - z) * n + z * h32[e];
      h32[e] = hn;
      h16[e] = (_Float16)hn;
      size_t idx = ((size_t)(B0 + lb) * T_ + t) * H_ + h;
      states_f[idx] = hn;
      states_h[idx] = (_Float16)hn;
    }
    __syncthreads();
  }
}

// ------------------------------------------------------------- TAM attention
// Block: one batch b, 8 query tiles (16 rows each). Flash-style over key pairs
// of 32, with double-buffered async DMA staging of K/V tiles into LDS.
// exp(tanh(s)) is bounded, so no running max; normalize at the end.
__global__ void __launch_bounds__(256) k_tam(
    const _Float16* __restrict__ states_h, float* __restrict__ tam) {
  __shared__ __align__(16) _Float16 Kt[2][32 * H_]; // 32 keys x 128, row-major
  __shared__ __align__(16) _Float16 Vt[H_ * 32];    // transposed: 128 x 32
  __shared__ __align__(16) _Float16 Pw[8][16 * 32];
  int tid = threadIdx.x, wave = tid >> 5, lane = tid & 31;
  int b = blockIdx.x >> 3;
  int qg = blockIdx.x & 7;
  int qi = qg * 8 + wave;                           // query tile (wave-uniform)
  const _Float16* base = states_h + (size_t)b * T_ * H_;
  int e0 = tid * 16;                                // this thread's tile chunk

  v16h qa[4];
#pragma unroll
  for (int s = 0; s < 4; ++s) qa[s] = load_frag(base + s * 32, H_, qi * 16);
  v8f zero = {};
  v8f acc[8];
#pragma unroll
  for (int n = 0; n < 8; ++n) acc[n] = zero;
  float dn[8];
#pragma unroll
  for (int r = 0; r < 8; ++r) dn[r] = 0.f;

  int npairs = qg * 4 + 4;                          // keys up to block's max i

  // preload tile 0 via async DMA
  async_copy32(&Kt[0][e0], base + e0);
  wait_async0();
  __syncthreads();

  for (int p = 0; p < npairs; ++p) {
    int c = p & 1;
    // kick off DMA of next tile into the other buffer; overlaps with compute
    if (p + 1 < npairs)
      async_copy32(&Kt[1 - c][e0], base + (size_t)((p + 1) * 32) * H_ + e0);

    // build transposed copy Vt from the landed tile (LDS -> reg -> LDS)
    {
      v8h lo = *(const v8h*)&Kt[c][e0];
      v8h hi = *(const v8h*)&Kt[c][e0 + 8];
#pragma unroll
      for (int i = 0; i < 8; ++i) {
        int e = e0 + i;
        Vt[(e & 127) * 32 + (e >> 7)] = lo[i];
        e = e0 + 8 + i;
        Vt[(e & 127) * 32 + (e >> 7)] = hi[i];
      }
    }
    __syncthreads();

    if (p * 32 < qi * 16 + 15) {                    // wave-uniform causal skip
#pragma unroll
      for (int q = 0; q < 2; ++q) {
        v8f s_acc = zero;
#pragma unroll
        for (int s = 0; s < 4; ++s) {
          v16h kb = load_frag(&Kt[c][0] + s * 32, H_, q * 16);
          s_acc = wmma_f16(qa[s], kb, s_acc);
        }
        int col = lane & 15, rbase = (lane >> 4) * 8;
        int jg = (p * 2 + q) * 16 + col;
#pragma unroll
        for (int r = 0; r < 8; ++r) {
          int ig = qi * 16 + rbase + r;
          float w = (jg < ig) ? __expf(tanhf(s_acc[r])) : 0.f;
          dn[r] += w;
          Pw[wave][(rbase + r) * 32 + q * 16 + col] = (_Float16)w;
        }
      }
      asm volatile("s_wait_dscnt 0" ::: "memory");  // wave-local LDS round trip
      v16h pa = load_frag(Pw[wave], 32, 0);
#pragma unroll
      for (int n = 0; n < 8; ++n) {
        v16h vb = load_frag(Vt, 32, n * 16);
        acc[n] = wmma_f16(pa, vb, acc[n]);
      }
    }
    if (p + 1 < npairs) wait_async0();              // next tile landed
    __syncthreads();
  }

  int col = lane & 15, rbase = (lane >> 4) * 8;
#pragma unroll
  for (int r = 0; r < 8; ++r) {
    float d = dn[r];
    d += __shfl_xor(d, 1);
    d += __shfl_xor(d, 2);
    d += __shfl_xor(d, 4);
    d += __shfl_xor(d, 8);
    float inv = (d > 0.f) ? 1.f / d : 0.f;
    int row = qi * 16 + rbase + r;
#pragma unroll
    for (int n = 0; n < 8; ++n)
      tam[((size_t)b * T_ + row) * H_ + n * 16 + col] = acc[n][r] * inv;
  }
}

// ------------------------------------- env = tanh(concat) @ W1^T + b1 ; fused W2
__global__ void __launch_bounds__(256) k_env(
    const float* __restrict__ states_f, const float* __restrict__ tam,
    const _Float16* __restrict__ W1h, const float* __restrict__ b1,
    const float* __restrict__ W2, const float* __restrict__ b2,
    float* __restrict__ env_out, float* __restrict__ out_flat) {
  const int LDA = 264;                               // padded, row stride 16B-aligned
  __shared__ __align__(16) _Float16 As[32 * LDA];
  __shared__ float outPart[32];
  int tid = threadIdx.x, wave = tid >> 5, lane = tid & 31;
  int m0 = blockIdx.x * 32;
  if (tid < 32) outPart[tid] = 0.f;
  // prefetch the W1 stream for this block's N-chunks
  __builtin_prefetch(W1h + (size_t)(tid & 3) * 128 * 2 * H_ + (tid >> 2) * 64, 0, 1);
  for (int e = tid; e < 32 * 256; e += 256) {
    int row = e >> 8, k = e & 255;
    size_t m = (size_t)m0 + row;
    float v = (k < H_) ? states_f[m * H_ + k] : tam[m * H_ + (k - H_)];
    As[row * LDA + k] = (_Float16)tanhf(v);
  }
  __syncthreads();

  int mi = wave >> 2, nj = wave & 3;                 // 2 M-tiles x 4 N-chunks
  v16h a[8];
#pragma unroll
  for (int s = 0; s < 8; ++s) a[s] = load_frag(As + s * 32, LDA, mi * 16);
  v8f zero = {};
  v8f acc[8];
#pragma unroll
  for (int n = 0; n < 8; ++n) acc[n] = zero;
#pragma unroll
  for (int n = 0; n < 8; ++n) {
    int ncol0 = nj * 128 + n * 16;
#pragma unroll
    for (int s = 0; s < 8; ++s) {                    // K = 256 = 8 * 32
      v16h bfrag = load_frag(W1h + s * 32, 2 * H_, ncol0);
      acc[n] = wmma_f16(a[s], bfrag, acc[n]);
    }
  }

  int colL = lane & 15, rbase = (lane >> 4) * 8;
  float part[8];
#pragma unroll
  for (int r = 0; r < 8; ++r) part[r] = 0.f;
#pragma unroll
  for (int n = 0; n < 8; ++n) {
    int col = nj * 128 + n * 16 + colL;
    float bias = b1[col], w2v = W2[col];
#pragma unroll
    for (int r = 0; r < 8; ++r) {
      size_t m = (size_t)m0 + mi * 16 + rbase + r;
      float e = acc[n][r] + bias;
      env_out[m * NENV + col] = e;
      part[r] += fmaxf(e, 0.f) * w2v;
    }
  }
#pragma unroll
  for (int r = 0; r < 8; ++r) {
    float s = part[r];
    s += __shfl_xor(s, 1);
    s += __shfl_xor(s, 2);
    s += __shfl_xor(s, 4);
    s += __shfl_xor(s, 8);
    if (colL == 0) atomicAdd(&outPart[mi * 16 + rbase + r], s);
  }
  __syncthreads();
  if (tid < 32) out_flat[m0 + tid] = outPart[tid] + b2[0];
}

// ------------------------------------------------------------------- h_last
__global__ void k_hlast(const float* __restrict__ states_f,
                        float* __restrict__ hlast) {
  int i = blockIdx.x * blockDim.x + threadIdx.x;
  if (i < B_ * H_) {
    int b = i / H_, h = i % H_;
    hlast[i] = states_f[((size_t)b * T_ + (T_ - 1)) * H_ + h];
  }
}

extern "C" void kernel_launch(void* const* d_in, const int* in_sizes, int n_in,
                              void* d_out, int out_size, void* d_ws, size_t ws_size,
                              hipStream_t stream) {
  const float* x    = (const float*)d_in[0];
  const float* h0   = (const float*)d_in[1];
  const float* W_ih = (const float*)d_in[2];
  const float* W_hh = (const float*)d_in[3];
  const float* b_ih = (const float*)d_in[4];
  const float* b_hh = (const float*)d_in[5];
  const float* W1   = (const float*)d_in[6];
  const float* b1   = (const float*)d_in[7];
  const float* W2   = (const float*)d_in[8];
  const float* b2   = (const float*)d_in[9];

  float* out_flat  = (float*)d_out;                        // (B*T)
  float* env_out   = out_flat + MTOT;                      // (B*T, 512)
  float* hlast_out = env_out + (size_t)MTOT * NENV;        // (B, H)

  char* ws = (char*)d_ws;
  size_t off = 0;
  auto carve = [&](size_t bytes) -> void* {
    void* p = (void*)(ws + off);
    off = (off + bytes + 255) & ~(size_t)255;
    return p;
  };
  _Float16* xh       = (_Float16*)carve((size_t)T_ * B_ * KIN * 2);
  _Float16* Wih_h    = (_Float16*)carve((size_t)G3 * KIN * 2);
  _Float16* Whh_h    = (_Float16*)carve((size_t)G3 * H_ * 2);
  _Float16* W1h      = (_Float16*)carve((size_t)NENV * 2 * H_ * 2);
  float*    Gi       = (float*)carve((size_t)MTOT * G3 * 4);
  float*    states_f = (float*)carve((size_t)MTOT * H_ * 4);
  _Float16* states_h = (_Float16*)carve((size_t)MTOT * H_ * 2);
  float*    tam      = (float*)carve((size_t)MTOT * H_ * 4);
  (void)ws_size; (void)in_sizes; (void)n_in; (void)out_size;

  k_pack<<<2048, 256, 0, stream>>>(x, W_ih, W_hh, W1, xh, Wih_h, Whh_h, W1h);
  k_gemm_gi<<<(MTOT / 16) * (G3 / 16) / 8, 256, 0, stream>>>(xh, Wih_h, b_ih, Gi);
  k_gru<<<B_ / 16, 256, 0, stream>>>(Whh_h, Gi, h0, b_hh, states_f, states_h);
  k_tam<<<B_ * (T_ / 128), 256, 0, stream>>>(states_h, tam);
  k_env<<<MTOT / 32, 256, 0, stream>>>(states_f, tam, W1h, b1, W2, b2, env_out,
                                       out_flat);
  k_hlast<<<(B_ * H_ + 255) / 256, 256, 0, stream>>>(states_f, hlast_out);
}
